// Net_15530601743096
// MI455X (gfx1250) — matrix operand order repaired and checked by reading, not compiled, not run on hardware
//
#include <hip/hip_runtime.h>
#include <hip/hip_bf16.h>
#include <math.h>

// ---------------------------------------------------------------------------
// 2-layer GCN for MI455X (gfx1250, wave32).
// GEMMs via v_wmma_f32_16x16x32_bf16; edge aggregation via hardware f32
// global atomics; self-loops folded into the epilogue kernels.
// ---------------------------------------------------------------------------

#define N_NODES 100000
#define N_EDGES 1600000
#define F_IN    128
#define F_HID   128
#define N_CLS   16

typedef __attribute__((ext_vector_type(16))) __bf16 v16bf;
typedef __attribute__((ext_vector_type(8)))  float  v8f;

static __device__ __forceinline__ float fatomic_add(float* p, float v) {
  return unsafeAtomicAdd(p, v);   // lowers to global_atomic_add_f32 on gfx1250
}

// ---------------- degree / normalization ----------------

__global__ void k_init_deg(float* deg) {
  int i = blockIdx.x * blockDim.x + threadIdx.x;
  if (i < N_NODES) deg[i] = 1.0f;        // self-loop contributes 1 to degree
}

__global__ void k_deg_accum(const int* __restrict__ ei, float* deg) {
  int e = blockIdx.x * blockDim.x + threadIdx.x;
  if (e < N_EDGES) fatomic_add(&deg[ei[N_EDGES + e]], 1.0f);  // dst row
}

__global__ void k_deg_to_dis(float* deg) {
  int i = blockIdx.x * blockDim.x + threadIdx.x;
  if (i < N_NODES) deg[i] = rsqrtf(deg[i]);   // deg >= 1 always
}

__global__ void k_zero(float* p, int n) {
  int i = blockIdx.x * blockDim.x + threadIdx.x;
  if (i < n) p[i] = 0.0f;
}

// ---------------- GEMM1: h = x(f32->bf16) @ W1, 128 -> 128 ----------------
// One wave computes a 16-row x 128-col tile. W1 is staged into LDS in exact
// per-lane B-fragment layout: fragment (kt,nt), lane l holds 16 bf16 where
//   N = nt*16 + (l&15),  K = kt*32 + 8*(l>>4) + e + 8*(e>>3)   (e = 0..15, K pairs/VGPR)
// A-fragment (16-bit A, 16x32): lane l, row M = l&15,
//   e in 0..7  -> K = kt*32 + 8*(l>>4) + e
//   e in 8..15 -> K = kt*32 + 8*(l>>4) + e + 8
// C/D (f32 16x16): lane l, vgpr v -> M = v + 8*(l>>4), N = l&15.

__global__ __launch_bounds__(256) void k_gemm1(const float* __restrict__ x,
                                               const float* __restrict__ W1,  // [128][128] row-major
                                               float* __restrict__ h) {
  __shared__ __bf16 lds_w[4 * 8 * 32 * 16];   // 32 KB, fragment-major
  for (int i = threadIdx.x; i < 4 * 8 * 32 * 16; i += 256) {
    int e    = i & 15;
    int lane = (i >> 4) & 31;
    int nt   = (i >> 9) & 7;
    int kt   = i >> 12;
    int k = kt * 32 + ((lane >> 4) << 3) + e + ((e >> 3) << 3);
    int n = nt * 16 + (lane & 15);
    lds_w[i] = (__bf16)W1[k * F_HID + n];
  }
  __syncthreads();

  const int wave = threadIdx.x >> 5;
  const int lane = threadIdx.x & 31;
  const int g    = lane >> 4;
  const int row0 = (blockIdx.x * 8 + wave) * 16;
  if (row0 >= N_NODES) return;                 // N_NODES % 16 == 0

  const float* xr = x + (size_t)(row0 + (lane & 15)) * F_IN;

  v8f acc[8];
  #pragma unroll
  for (int nt = 0; nt < 8; ++nt) acc[nt] = {};

  const v16bf* wf = (const v16bf*)lds_w;
  #pragma unroll
  for (int kt = 0; kt < 4; ++kt) {
    const int kb = kt * 32 + g * 8;
    v16bf a;
    #pragma unroll
    for (int e = 0; e < 8; ++e)  a[e] = (__bf16)xr[kb + e];
    #pragma unroll
    for (int e = 8; e < 16; ++e) a[e] = (__bf16)xr[kb + 8 + e];
    #pragma unroll
    for (int nt = 0; nt < 8; ++nt) {
      v16bf b = wf[(kt * 8 + nt) * 32 + lane];
      acc[nt] = __builtin_amdgcn_wmma_f32_16x16x32_bf16(
          false, a, false, b, (short)0, acc[nt], false, false);
    }
  }

  float* hr = h + (size_t)row0 * F_HID;
  #pragma unroll
  for (int nt = 0; nt < 8; ++nt) {
    #pragma unroll
    for (int v = 0; v < 8; ++v) {
      hr[(size_t)(v + 8 * g) * F_HID + nt * 16 + (lane & 15)] = acc[nt][v];
    }
  }
}

// ---------------- edge scatter, layer 1: one wave per edge ----------------

__global__ __launch_bounds__(256) void k_scatter1(const int* __restrict__ ei,
                                                  const float* __restrict__ h,
                                                  const float* __restrict__ dis,
                                                  float* __restrict__ agg) {
  int t = blockIdx.x * 256 + threadIdx.x;
  int e = t >> 5;
  if (e >= N_EDGES) return;
  int lane = t & 31;
  int s = ei[e], d = ei[N_EDGES + e];
  float w = dis[s] * dis[d];
  float4 v = ((const float4*)(h + (size_t)s * F_HID))[lane];
  float* ad = agg + (size_t)d * F_HID + lane * 4;
  fatomic_add(ad + 0, v.x * w);
  fatomic_add(ad + 1, v.y * w);
  fatomic_add(ad + 2, v.z * w);
  fatomic_add(ad + 3, v.w * w);
}

// ---- epilogue 1: + self-loop + bias, ReLU, cast to bf16 for next GEMM ----

__global__ void k_relu1(const float* __restrict__ agg, const float* __restrict__ h,
                        const float* __restrict__ dis, const float* __restrict__ b1,
                        __bf16* __restrict__ hout) {
  int t = blockIdx.x * blockDim.x + threadIdx.x;   // t = n*128 + f
  if (t >= N_NODES * F_HID) return;
  int n = t >> 7, f = t & 127;
  float di = dis[n];
  float v  = agg[t] + h[t] * di * di + b1[f];
  hout[t]  = (__bf16)fmaxf(v, 0.0f);
}

// ---------------- GEMM2: h2 = h1(bf16) @ W2, 128 -> 16 ----------------

__global__ __launch_bounds__(256) void k_gemm2(const __bf16* __restrict__ hin,
                                               const float* __restrict__ W2,  // [128][16]
                                               float* __restrict__ h2) {
  __shared__ __bf16 lds_w[4 * 32 * 16];   // 4 KB
  for (int i = threadIdx.x; i < 4 * 32 * 16; i += 256) {
    int e    = i & 15;
    int lane = (i >> 4) & 31;
    int kt   = i >> 9;
    int k = kt * 32 + ((lane >> 4) << 3) + e + ((e >> 3) << 3);
    lds_w[i] = (__bf16)W2[k * N_CLS + (lane & 15)];
  }
  __syncthreads();

  const int wave = threadIdx.x >> 5;
  const int lane = threadIdx.x & 31;
  const int g    = lane >> 4;
  const int row0 = (blockIdx.x * 8 + wave) * 16;
  if (row0 >= N_NODES) return;

  const __bf16* hr = hin + (size_t)(row0 + (lane & 15)) * F_HID;
  const v16bf* wf = (const v16bf*)lds_w;
  v8f acc = {};
  #pragma unroll
  for (int kt = 0; kt < 4; ++kt) {
    const int kb = kt * 32 + g * 8;
    v16bf a;
    #pragma unroll
    for (int e = 0; e < 8; ++e)  a[e] = hr[kb + e];
    #pragma unroll
    for (int e = 8; e < 16; ++e) a[e] = hr[kb + 8 + e];
    v16bf b = wf[kt * 32 + lane];
    acc = __builtin_amdgcn_wmma_f32_16x16x32_bf16(
        false, a, false, b, (short)0, acc, false, false);
  }
  float* out = h2 + (size_t)row0 * N_CLS;
  #pragma unroll
  for (int v = 0; v < 8; ++v)
    out[(size_t)(v + 8 * g) * N_CLS + (lane & 15)] = acc[v];
}

// ---------------- edge scatter, layer 2: 8 lanes per edge ----------------

__global__ __launch_bounds__(256) void k_scatter2(const int* __restrict__ ei,
                                                  const float* __restrict__ h2,
                                                  const float* __restrict__ dis,
                                                  float* __restrict__ agg2) {
  int t = blockIdx.x * 256 + threadIdx.x;
  int e = t >> 3;
  if (e >= N_EDGES) return;
  int p = t & 7;
  int s = ei[e], d = ei[N_EDGES + e];
  float w = dis[s] * dis[d];
  float2 v = ((const float2*)(h2 + (size_t)s * N_CLS))[p];
  float* ad = agg2 + (size_t)d * N_CLS + p * 2;
  fatomic_add(ad + 0, v.x * w);
  fatomic_add(ad + 1, v.y * w);
}

// ---- epilogue 2: + self-loop + bias, log_softmax over 16 classes ----

__global__ void k_final(const float* __restrict__ agg2, const float* __restrict__ h2,
                        const float* __restrict__ dis, const float* __restrict__ b2,
                        float* __restrict__ out) {
  int n = blockIdx.x * blockDim.x + threadIdx.x;
  if (n >= N_NODES) return;
  float dd = dis[n] * dis[n];
  float v[N_CLS];
  float mx = -INFINITY;
  #pragma unroll
  for (int c = 0; c < N_CLS; ++c) {
    v[c] = agg2[(size_t)n * N_CLS + c] + h2[(size_t)n * N_CLS + c] * dd + b2[c];
    mx = fmaxf(mx, v[c]);
  }
  float s = 0.0f;
  #pragma unroll
  for (int c = 0; c < N_CLS; ++c) s += expf(v[c] - mx);
  float lse = mx + logf(s);
  #pragma unroll
  for (int c = 0; c < N_CLS; ++c) out[(size_t)n * N_CLS + c] = v[c] - lse;
}

// ---------------------------------------------------------------------------

extern "C" void kernel_launch(void* const* d_in, const int* in_sizes, int n_in,
                              void* d_out, int out_size, void* d_ws, size_t ws_size,
                              hipStream_t stream) {
  const float* x  = (const float*)d_in[0];
  const int*   ei = (const int*)d_in[1];   // [2][E] int32 (JAX x64 disabled)
  const float* W1 = (const float*)d_in[2];
  const float* b1 = (const float*)d_in[3];
  const float* W2 = (const float*)d_in[4];
  const float* b2 = (const float*)d_in[5];
  float* out = (float*)d_out;

  // Workspace layout (bytes), 256-aligned regions; h2/agg2 overlay h1/agg1
  // once those are dead.
  char* ws = (char*)d_ws;
  size_t off = 0;
  float*  dis  = (float*)(ws + off); off += ((size_t)N_NODES * 4 + 255) & ~(size_t)255;
  float*  h1   = (float*)(ws + off); off += (size_t)N_NODES * F_HID * 4;
  float*  agg1 = (float*)(ws + off); off += (size_t)N_NODES * F_HID * 4;
  __bf16* h1r  = (__bf16*)(ws + off); off += (size_t)N_NODES * F_HID * 2;
  float*  h2   = h1;     // reuse after k_relu1
  float*  agg2 = agg1;   // reuse after k_relu1

  const int B = 256;
  const int gN     = (N_NODES + B - 1) / B;             // per-node
  const int gE     = (N_EDGES + B - 1) / B;             // per-edge
  const int gNF    = (N_NODES * F_HID + B - 1) / B;     // per node-feature
  const int gGemm  = ((N_NODES / 16) + 7) / 8;          // 8 waves x 16 rows / block
  const int gScat1 = ((N_EDGES * 32) + B - 1) / B;      // wave per edge
  const int gScat2 = ((N_EDGES * 8) + B - 1) / B;       // 8 lanes per edge

  // Normalization: deg (with self loop) -> dis = rsqrt(deg), in place.
  k_init_deg <<<gN, B, 0, stream>>>(dis);
  k_deg_accum<<<gE, B, 0, stream>>>(ei, dis);
  k_deg_to_dis<<<gN, B, 0, stream>>>(dis);

  // Layer 1.
  k_gemm1   <<<gGemm, B, 0, stream>>>(x, W1, h1);
  k_zero    <<<gNF, B, 0, stream>>>(agg1, N_NODES * F_HID);
  k_scatter1<<<gScat1, B, 0, stream>>>(ei, h1, dis, agg1);
  k_relu1   <<<gNF, B, 0, stream>>>(agg1, h1, dis, b1, h1r);

  // Layer 2 (h2/agg2 overlay h1/agg1).
  k_gemm2   <<<gGemm, B, 0, stream>>>(h1r, W2, h2);
  k_zero    <<<(N_NODES * N_CLS + B - 1) / B, B, 0, stream>>>(agg2, N_NODES * N_CLS);
  k_scatter2<<<gScat2, B, 0, stream>>>(ei, h2, dis, agg2);
  k_final   <<<gN, B, 0, stream>>>(agg2, h2, dis, b2, out);
}
